// OCR_36361193128060
// MI455X (gfx1250) — compile-verified, matrix operand order checked by reading, not run on hardware
//
#include <hip/hip_runtime.h>

// ---------------- CDNA5 WMMA types ----------------
typedef _Float16 __attribute__((ext_vector_type(16))) v16h;
typedef float    __attribute__((ext_vector_type(8)))  v8f;

#define NB   128   // batch / hypotheses
#define DM   512   // model dim
#define NH   8     // heads
#define DHD  64    // head dim
#define DFF  2048
#define TCACHE 256
#define SMEM 512
#define TSTRIDE 512  // scores/probs stride per (n,h)
#define DSPLIT 4     // d-dimension split for the ctx streaming pass

// Load a 16-wide K-contiguous fragment (A operand, or B operand when the
// "B" matrix is stored row-major along K, i.e. x @ W^T form).
// Layout per ISA 7.12.2 (16-bit A 16x32): lanes0-15 K in {0..7,16..23},
// lanes16-31 K in {8..15,24..31}; element i<8 -> K=half*8+i, i>=8 -> +16.
__device__ inline v16h load_frag_k(const float* __restrict__ row, int kOff, int half) {
    const float4* p = (const float4*)(row + kOff + half * 8);
    float4 f0 = p[0];
    float4 f1 = p[1];
    const float4* q = (const float4*)(row + kOff + 16 + half * 8);
    float4 f2 = q[0];
    float4 f3 = q[1];
    v16h a;
    a[0]=(_Float16)f0.x; a[1]=(_Float16)f0.y; a[2]=(_Float16)f0.z; a[3]=(_Float16)f0.w;
    a[4]=(_Float16)f1.x; a[5]=(_Float16)f1.y; a[6]=(_Float16)f1.z; a[7]=(_Float16)f1.w;
    a[8]=(_Float16)f2.x; a[9]=(_Float16)f2.y; a[10]=(_Float16)f2.z; a[11]=(_Float16)f2.w;
    a[12]=(_Float16)f3.x; a[13]=(_Float16)f3.y; a[14]=(_Float16)f3.z; a[15]=(_Float16)f3.w;
    return a;
}

// B operand when B is stored [K][N] row-major along N (C = A @ B, "NN" form).
__device__ inline v16h load_frag_nn(const float* __restrict__ B, long long ldb,
                                    int ncol, int kOff, int half) {
    v16h b;
#pragma unroll
    for (int j = 0; j < 8; ++j) {
        b[j]     = (_Float16)B[(size_t)(kOff + half * 8 + j) * ldb + ncol];
        b[8 + j] = (_Float16)B[(size_t)(kOff + 16 + half * 8 + j) * ldb + ncol];
    }
    return b;
}

// Generic tiled GEMM: C[z] = act(alpha * A[z] @ op(B[z]) + bias[z])
// BT=1: C[m][n] = sum_k A[m][k] * B[n][k]   (x @ W^T, W rows contiguous in K)
// BT=0: C[m][n] = sum_k A[m][k] * B[k][n]
// One wave (32 lanes) per 16x16 output tile. M,N multiples of 16; K of 32.
template <int BT>
__global__ __launch_bounds__(32) void gemm16_wmma(
    const float* __restrict__ A, long long lda, long long Abs,
    const float* __restrict__ B, long long ldb, long long Bbs,
    const float* __restrict__ bias, long long biasBs,
    float* __restrict__ C, long long ldc, long long Cbs,
    int K, float alpha, int relu) {
    int lane = threadIdx.x;
    int half = lane >> 4;
    int mn   = lane & 15;
    int n0 = blockIdx.x * 16;
    int m0 = blockIdx.y * 16;
    int z  = blockIdx.z;
    A += (size_t)z * Abs;
    B += (size_t)z * Bbs;
    C += (size_t)z * Cbs;
    if (bias) bias += (size_t)z * biasBs;

    const float* arow = A + (size_t)(m0 + mn) * lda;
    const float* brow = B + (size_t)(n0 + mn) * ldb;  // used only when BT=1

    v8f c = {};
    for (int kk = 0; kk < K; kk += 32) {
        v16h a = load_frag_k(arow, kk, half);
        v16h b = BT ? load_frag_k(brow, kk, half)
                    : load_frag_nn(B, ldb, n0 + mn, kk, half);
        c = __builtin_amdgcn_wmma_f32_16x16x32_f16(false, a, false, b, (short)0, c,
                                                   false, false);
    }
    int nn = n0 + mn;
    int mbase = m0 + (half ? 8 : 0);
    float bv = bias ? bias[nn] : 0.f;
#pragma unroll
    for (int r = 0; r < 8; ++r) {
        float v = alpha * c[r] + bv;
        if (relu) v = v > 0.f ? v : 0.f;
        C[(size_t)(mbase + r) * ldc + nn] = v;
    }
}

// Embedding gather + sinusoidal positional encoding at position *offp.
__global__ void embed_pe_k(const int* __restrict__ toks,
                           const float* __restrict__ embd,
                           const int* __restrict__ offp,
                           float* __restrict__ tgt) {
    int idx = blockIdx.x * 256 + threadIdx.x;
    if (idx >= NB * DM) return;
    int n = idx >> 9;
    int d = idx & (DM - 1);
    float off = (float)(*offp);
    int i2 = d & ~1;
    float f = __expf((float)i2 * (-9.210340371976184f / (float)DM));  // -ln(1e4)/D
    float ang = off * f;
    float pe = (d & 1) ? cosf(ang) : sinf(ang);
    tgt[idx] = embd[(size_t)toks[n] * DM + d] + pe;
}

// qb[n][h] = (1/8) * q[n,h,:] . bk[h*64:...]
__global__ void qdotb_k(const float* __restrict__ q, const float* __restrict__ bk,
                        float* __restrict__ qb) {
    int idx = blockIdx.x * blockDim.x + threadIdx.x;
    if (idx >= NB * NH) return;
    int n = idx >> 3, h = idx & 7;
    const float* qq = q + (size_t)n * DM + h * DHD;
    const float* bb = bk + h * DHD;
    float s = 0.f;
#pragma unroll 8
    for (int j = 0; j < DHD; ++j) s += qq[j] * bb[j];
    qb[idx] = 0.125f * s;
}

// scores[n][h][t] = qeff[n][h] . x_t(n) + qb[n][h]  (mask -> -1e9)
// One wave per (16 t-rows, n); 16 head-columns (8 valid) via WMMA.
// Columns h>=8 read a clamped qeff row and compute dead values that are
// never stored: keeps the inner loop branch-free with EXEC all-ones.
__global__ __launch_bounds__(32) void attn_scores_k(
    const float* __restrict__ xc, const float* __restrict__ x2,
    const float* __restrict__ qeff, const float* __restrict__ qb,
    const unsigned char* __restrict__ mask, float* __restrict__ scores,
    int Tk, int split) {
    int n = blockIdx.y;
    int t0 = blockIdx.x * 16;
    int lane = threadIdx.x;
    int half = lane >> 4;
    int m = lane & 15;

    int t = t0 + m;
    int tc = t < Tk ? t : Tk - 1;
    const float* arow = (tc < split) ? xc + ((size_t)tc * NB + n) * DM
                                     : x2 + (size_t)n * DM;
    int h = m;
    int hq = h < 7 ? h : 7;  // clamp: dead columns read row 7 (safe memory)
    const float* brow = qeff + ((size_t)n * NH + hq) * DM;

    v8f c = {};
    for (int kk = 0; kk < DM; kk += 32) {
        v16h a = load_frag_k(arow, kk, half);
        v16h b = load_frag_k(brow, kk, half);
        c = __builtin_amdgcn_wmma_f32_16x16x32_f16(false, a, false, b, (short)0, c,
                                                   false, false);
    }
    if (h >= NH) return;
    float qbv = qb[n * NH + h];
#pragma unroll
    for (int r = 0; r < 8; ++r) {
        int tt = t0 + r + (half ? 8 : 0);
        if (tt < Tk) {
            float s = c[r] + qbv;
            if (mask && mask[(size_t)n * Tk + tt]) s = -1e9f;
            scores[((size_t)n * NH + h) * TSTRIDE + tt] = s;
        }
    }
}

// In-place softmax over Tk for each (n,h); one wave each.
__global__ __launch_bounds__(32) void softmax_k(float* __restrict__ scores, int Tk) {
    int id = blockIdx.x;
    float* s = scores + (size_t)id * TSTRIDE;
    int lane = threadIdx.x;
    float m = -3.4e38f;
    for (int t = lane; t < Tk; t += 32) m = fmaxf(m, s[t]);
#pragma unroll
    for (int o = 16; o; o >>= 1) m = fmaxf(m, __shfl_xor(m, o, 32));
    float sum = 0.f;
    for (int t = lane; t < Tk; t += 32) {
        float e = __expf(s[t] - m);
        s[t] = e;
        sum += e;
    }
#pragma unroll
    for (int o = 16; o; o >>= 1) sum += __shfl_xor(sum, o, 32);
    float r = 1.f / sum;
    for (int t = lane; t < Tk; t += 32) s[t] *= r;
}

// ctx[n][h][dslice] = sum_t probs[n][h][t] * x_t(n)[dslice]
// Grid (NB, DSPLIT); one wave per block, each owning 128 d-values.
// This is the main HBM-streaming pass; DSPLIT raises wave concurrency.
__global__ __launch_bounds__(32) void ctx_k(const float* __restrict__ xc,
                                            const float* __restrict__ x2,
                                            const float* __restrict__ probs,
                                            float* __restrict__ ctx,
                                            int Tk, int split) {
    __shared__ float sp[NH * TSTRIDE];
    int n = blockIdx.x;
    int dbase = blockIdx.y * (DM / DSPLIT);  // 128 d-values per block
    int tid = threadIdx.x;
    for (int i = tid; i < NH * TSTRIDE; i += 32) {
        int t = i & (TSTRIDE - 1);
        int h = i >> 9;
        sp[i] = (t < Tk) ? probs[((size_t)n * NH + h) * TSTRIDE + t] : 0.f;
    }
    __syncthreads();

    float acc[NH][4] = {};
    for (int t = 0; t < Tk; ++t) {
        const float* row = ((t < split) ? xc + ((size_t)t * NB + n) * DM
                                        : x2 + (size_t)n * DM) + dbase;
        if (t + 16 < split) {
            const float* rp = xc + ((size_t)(t + 16) * NB + n) * DM + dbase;
            __builtin_prefetch(rp + tid * 4, 0, 1);
        }
        float4 xv = ((const float4*)row)[tid];
#pragma unroll
        for (int h = 0; h < NH; ++h) {
            float p = sp[(h << 9) + t];
            acc[h][0] += p * xv.x;
            acc[h][1] += p * xv.y;
            acc[h][2] += p * xv.z;
            acc[h][3] += p * xv.w;
        }
    }
#pragma unroll
    for (int h = 0; h < NH; ++h) {
        float* o = ctx + ((size_t)n * NH + h) * DM + dbase + tid * 4;
        o[0] = acc[h][0]; o[1] = acc[h][1]; o[2] = acc[h][2]; o[3] = acc[h][3];
    }
}

// tgt[n] = LayerNorm(tgt[n] + addv[n]) * w + b   (one block per row)
__global__ __launch_bounds__(128) void add_ln_k(float* __restrict__ tgt,
                                                const float* __restrict__ addv,
                                                const float* __restrict__ w,
                                                const float* __restrict__ b) {
    __shared__ float red[128];
    int n = blockIdx.x, tid = threadIdx.x;
    size_t base = (size_t)n * DM + tid * 4;
    float v[4];
    float s = 0.f;
#pragma unroll
    for (int j = 0; j < 4; ++j) {
        v[j] = tgt[base + j] + addv[base + j];
        s += v[j];
    }
    red[tid] = s;
    __syncthreads();
    for (int o = 64; o; o >>= 1) {
        if (tid < o) red[tid] += red[tid + o];
        __syncthreads();
    }
    float mean = red[0] * (1.f / DM);
    __syncthreads();
    float s2 = 0.f;
#pragma unroll
    for (int j = 0; j < 4; ++j) {
        v[j] -= mean;
        s2 += v[j] * v[j];
    }
    red[tid] = s2;
    __syncthreads();
    for (int o = 64; o; o >>= 1) {
        if (tid < o) red[tid] += red[tid + o];
        __syncthreads();
    }
    float rstd = rsqrtf(red[0] * (1.f / DM) + 1e-5f);
#pragma unroll
    for (int j = 0; j < 4; ++j)
        tgt[base + j] = v[j] * rstd * w[tid * 4 + j] + b[tid * 4 + j];
}

extern "C" void kernel_launch(void* const* d_in, const int* in_sizes, int n_in,
                              void* d_out, int out_size, void* d_ws, size_t ws_size,
                              hipStream_t stream) {
    const int*   last_toks = (const int*)d_in[0];
    const float* cached    = (const float*)d_in[1];
    const float* memory    = (const float*)d_in[2];
    const unsigned char* memory_mask = (const unsigned char*)d_in[3];
    const float* embd   = (const float*)d_in[4];
    const float* Wqkv_s = (const float*)d_in[5];
    const float* bqkv_s = (const float*)d_in[6];
    const float* Wo_s   = (const float*)d_in[7];
    const float* bo_s   = (const float*)d_in[8];
    const float* Wqkv_c = (const float*)d_in[9];
    const float* bqkv_c = (const float*)d_in[10];
    const float* Wo_c   = (const float*)d_in[11];
    const float* bo_c   = (const float*)d_in[12];
    const float* W1 = (const float*)d_in[13];
    const float* b1 = (const float*)d_in[14];
    const float* W2 = (const float*)d_in[15];
    const float* b2 = (const float*)d_in[16];
    const float* ln1_w = (const float*)d_in[17];
    const float* ln1_b = (const float*)d_in[18];
    const float* ln2_w = (const float*)d_in[19];
    const float* ln2_b = (const float*)d_in[20];
    const float* ln3_w = (const float*)d_in[21];
    const float* ln3_b = (const float*)d_in[22];
    const int*   offset = (const int*)d_in[23];

    float* tgt = (float*)d_out;  // running activation [N, D]

    // workspace carve-up (floats)
    float* w0      = (float*)d_ws;
    float* qbuf    = w0;                 // 128*512
    float* qeff    = qbuf + NB * DM;     // 128*8*512
    float* qb      = qeff + NB * NH * DM;      // 1024
    float* scores  = qb + NB * NH;             // 128*8*512
    float* ctx     = scores + NB * NH * TSTRIDE;  // 128*8*512
    float* attnbuf = ctx + NB * NH * DM;   // 128*512
    float* obuf    = attnbuf + NB * DM;    // 128*512
    float* ffh     = obuf + NB * DM;       // 128*2048

    const long long DD = (long long)DM * DM;

    // tgt = embd[toks] + PE(offset)
    embed_pe_k<<<(NB * DM + 255) / 256, 256, 0, stream>>>(last_toks, embd, offset, tgt);

    auto attn_block = [&](const float* WL, const float* bL, const float* WoL,
                          const float* boL, const float* xc, const float* x2,
                          int Tk, int split, const unsigned char* mask,
                          const float* lnw, const float* lnb) {
        // q = tgt @ Wq^T + bq
        gemm16_wmma<1><<<dim3(DM / 16, NB / 16, 1), 32, 0, stream>>>(
            tgt, DM, 0, WL, DM, 0, bL, 0, qbuf, DM, 0, DM, 1.f, 0);
        // qeff[h] = (1/8) * Q_h @ Wk_h   (fold K-projection into query)
        gemm16_wmma<0><<<dim3(DM / 16, NB / 16, NH), 32, 0, stream>>>(
            qbuf, DM, DHD, WL + DD, DM, (long long)DHD * DM, nullptr, 0,
            qeff, (long long)NH * DM, DM, DHD, 0.125f, 0);
        // qb[n][h] = (1/8) q . bk
        qdotb_k<<<(NB * NH + 127) / 128, 128, 0, stream>>>(qbuf, bL + DM, qb);
        // scores over raw kv inputs
        attn_scores_k<<<dim3((Tk + 15) / 16, NB), 32, 0, stream>>>(
            xc, x2, qeff, qb, mask, scores, Tk, split);
        softmax_k<<<NB * NH, 32, 0, stream>>>(scores, Tk);
        // weighted context of raw kv inputs
        ctx_k<<<dim3(NB, DSPLIT), 32, 0, stream>>>(xc, x2, scores, ctx, Tk, split);
        // o_h = ctx_h @ Wv_h^T + bv_h   (fold V-projection after softmax)
        gemm16_wmma<1><<<dim3(DHD / 16, NB / 16, NH), 32, 0, stream>>>(
            ctx, (long long)NH * DM, DM, WL + 2 * DD, DM, (long long)DHD * DM,
            bL + 2 * DM, DHD, attnbuf, DM, DHD, DM, 1.f, 0);
        // out = o @ Wo^T + bo
        gemm16_wmma<1><<<dim3(DM / 16, NB / 16, 1), 32, 0, stream>>>(
            attnbuf, DM, 0, WoL, DM, 0, boL, 0, obuf, DM, 0, DM, 1.f, 0);
        // tgt = LN(tgt + out)
        add_ln_k<<<NB, 128, 0, stream>>>(tgt, obuf, lnw, lnb);
    };

    for (int l = 0; l < 6; ++l) {
        const float* WLs = Wqkv_s + (size_t)l * 3 * DD;
        const float* bLs = bqkv_s + (size_t)l * 3 * DM;
        const float* WLc = Wqkv_c + (size_t)l * 3 * DD;
        const float* bLc = bqkv_c + (size_t)l * 3 * DM;
        const float* xc  = cached + (size_t)l * TCACHE * NB * DM;

        // self-attention: kv = [cached_l ; tgt], Tk = T+1
        attn_block(WLs, bLs, Wo_s + (size_t)l * DD, bo_s + (size_t)l * DM,
                   xc, tgt, TCACHE + 1, TCACHE, nullptr,
                   ln1_w + (size_t)l * DM, ln1_b + (size_t)l * DM);
        // cross-attention: kv = memory, Tk = S, with key padding mask
        attn_block(WLc, bLc, Wo_c + (size_t)l * DD, bo_c + (size_t)l * DM,
                   memory, tgt, SMEM, SMEM, memory_mask,
                   ln2_w + (size_t)l * DM, ln2_b + (size_t)l * DM);
        // FFN
        gemm16_wmma<1><<<dim3(DFF / 16, NB / 16, 1), 32, 0, stream>>>(
            tgt, DM, 0, W1 + (size_t)l * DFF * DM, DM, 0, b1 + (size_t)l * DFF, 0,
            ffh, DFF, 0, DM, 1.f, 1);
        gemm16_wmma<1><<<dim3(DM / 16, NB / 16, 1), 32, 0, stream>>>(
            ffh, DFF, 0, W2 + (size_t)l * DM * DFF, DFF, 0, b2 + (size_t)l * DM, 0,
            obuf, DM, 0, DFF, 1.f, 0);
        add_ln_k<<<NB, 128, 0, stream>>>(tgt, obuf,
                                         ln3_w + (size_t)l * DM, ln3_b + (size_t)l * DM);
    }
}
